// StokenAttention_55791625175252
// MI455X (gfx1250) — matrix operand order repaired and checked by reading, not compiled, not run on hardware
//
#include <hip/hip_runtime.h>

typedef _Float16 v16h __attribute__((ext_vector_type(16)));
typedef _Float16 v8h  __attribute__((ext_vector_type(8)));
typedef float    v8f  __attribute__((ext_vector_type(8)));

namespace {

constexpr int BB = 8, CC = 256, HH = 16, WW = 16;   // batch, channels, token grid
constexpr int NTOK = HH * WW;                       // 256 tokens
constexpr int NHEAD = 8;
constexpr float ATTN_SC = 0.17677669529663687f;     // 32^-0.5
constexpr float AFF_SC  = 0.0625f;                  // 256^-0.5

// ---------------- fragment helpers (gfx1250 wave32 WMMA f16 layouts) -------
__device__ __forceinline__ v8f vzero8f() {
  v8f r;
#pragma unroll
  for (int i = 0; i < 8; ++i) r[i] = 0.f;
  return r;
}
__device__ __forceinline__ v16h vzero16h() {
  v16h r;
#pragma unroll
  for (int i = 0; i < 16; ++i) r[i] = (_Float16)0;
  return r;
}
__device__ __forceinline__ v16h pack16(v8h lo, v8h hi) {
  v16h r;
#pragma unroll
  for (int i = 0; i < 8; ++i) { r[i] = lo[i]; r[i + 8] = hi[i]; }
  return r;
}
// A (16x32 f16) fragment: row-major src, base already at (m0,k0).
// Lane m<16: K = 0..15 ; lane>=16: K = 16..31.
__device__ __forceinline__ v16h load_a(const _Float16* base, int ld, int lane) {
  const _Float16* p = base + (lane & 15) * ld + ((lane & 16) ? 16 : 0);
  return pack16(*(const v8h*)p, *(const v8h*)(p + 8));
}
// B (32x16 f16) fragment from Bt (N x K row-major), base at (n0,k0).
// Lane n<16 holds K=0..15, lane>=16 holds K=16..31 of column n.
__device__ __forceinline__ v16h load_bt(const _Float16* base, int ld, int lane) {
  const _Float16* p = base + (lane & 15) * ld + ((lane & 16) ? 16 : 0);
  return pack16(*(const v8h*)p, *(const v8h*)(p + 8));
}
__device__ __forceinline__ v8f wmma16(v16h a, v16h b, v8f c) {
  return __builtin_amdgcn_wmma_f32_16x16x32_f16(false, a, false, b, (short)0, c,
                                                false, false);
}

// ---------------- K0: cast weights to f16 ---------------------------------
__global__ void k_cvt_w(const float* wqkv, const float* wproj,
                        _Float16* wqkv_h, _Float16* wproj_h) {
  int i = blockIdx.x * 256 + threadIdx.x;
  if (i < 768 * 256) wqkv_h[i] = (_Float16)wqkv[i];
  if (i < 256 * 256) wproj_h[i] = (_Float16)wproj[i];
}

// ---------------- K1: 8x8 average pool -> stoken (b,hh,ww,c) f16 ----------
__global__ void __launch_bounds__(128) k_pool(const float* x, _Float16* stoken_hwc) {
  int blk = blockIdx.x;                 // b*C*HH + c*HH + hh
  int hh = blk & 15;
  int c  = (blk >> 4) & 255;
  int b  = blk >> 12;
  int w  = threadIdx.x;                 // 0..127
  const float* p = x + ((((size_t)b * CC + c) * 128) + hh * 8) * 128 + w;
  float acc = 0.f;
#pragma unroll
  for (int r = 0; r < 8; ++r) acc += p[r * 128];
  for (int off = 1; off < 8; off <<= 1) acc += __shfl_xor(acc, off, 8);
  if ((w & 7) == 0)
    stoken_hwc[(((size_t)b * 16 + hh) * 16 + (w >> 3)) * CC + c] =
        (_Float16)(acc * (1.0f / 64.0f));
}

// ---------------- K2: affinity softmax + super-token aggregation ----------
// per block: one (b, token). logits(64x9)=pix(64x256)@sf9 ; softmax over 9 ;
// sfk(9x256)=affT(9x64)@pix(64x256). All matmuls on WMMA.
__global__ void __launch_bounds__(128) k_aff(const float* x,
                                             const _Float16* stoken_hwc,
                                             _Float16* aff_g, float* asum_pre,
                                             float* sfk) {
  __shared__ __align__(16) _Float16 pix_s[64][256];   // (p, c)
  __shared__ __align__(16) _Float16 pixT_s[256][64];  // (c, p)
  __shared__ float logit_s[64][16];
  __shared__ __align__(16) _Float16 affT_s[16][64];   // (j, p), rows 9..15 zero

  int blk = blockIdx.x;
  int n = blk & 255, b = blk >> 8;
  int yy = n >> 4, xx = n & 15;
  int tid = threadIdx.x, lane = tid & 31, wv = tid >> 5;

  // phase 0: gather 8x8 pixel patch (f32->f16), both layouts
  for (int cc = 0; cc < 2; ++cc) {
    int c = tid + cc * 128;
    const float* px = x + ((((size_t)b * CC + c) * 128) + yy * 8) * 128 + xx * 8;
#pragma unroll
    for (int r = 0; r < 8; ++r) {
      float4 lo = *(const float4*)(px + r * 128);
      float4 hi = *(const float4*)(px + r * 128 + 4);
      float vals[8] = {lo.x, lo.y, lo.z, lo.w, hi.x, hi.y, hi.z, hi.w};
#pragma unroll
      for (int s = 0; s < 8; ++s) {
        _Float16 hv = (_Float16)vals[s];
        pix_s[r * 8 + s][c] = hv;
        pixT_s[c][r * 8 + s] = hv;
      }
    }
  }
  __syncthreads();

  // phase 1: logits = pix @ sf9  (M=64 -> 4 waves x 1 tile, K=256, N=9 of 16)
  {
    int m0 = wv * 16;
    int j = lane & 15;
    int dy = j / 3, dx = j - dy * 3;
    int y2 = yy + dy - 1, x2 = xx + dx - 1;
    bool bval = (j < 9) && ((unsigned)y2 < 16u) && ((unsigned)x2 < 16u);
    const _Float16* brow =
        stoken_hwc + ((long)(b * 16 + y2) * 16 + x2) * CC;  // deref only if bval
    int kb2 = (lane & 16) ? 16 : 0;
    v8f acc = vzero8f();
#pragma unroll
    for (int k0 = 0; k0 < 256; k0 += 32) {
      v16h a = load_a(&pix_s[m0][0] + k0, 256, lane);
      v16h bm;
      if (bval) {
        const _Float16* p = brow + k0 + kb2;
        bm = pack16(*(const v8h*)p, *(const v8h*)(p + 8));
      } else {
        bm = vzero16h();
      }
      acc = wmma16(a, bm, acc);
    }
    int col = lane & 15;
    int rb = m0 + ((lane >> 4) << 3);
#pragma unroll
    for (int i = 0; i < 8; ++i) logit_s[rb + i][col] = acc[i];
  }
  __syncthreads();

  // phase 2: softmax over 9 per pixel row; emit aff (p,16) global + affT LDS
  if (tid < 64) {
    float v[9];
    float m = -1e30f;
#pragma unroll
    for (int j = 0; j < 9; ++j) {
      v[j] = logit_s[tid][j] * AFF_SC;
      m = fmaxf(m, v[j]);
    }
    float s = 0.f;
#pragma unroll
    for (int j = 0; j < 9; ++j) {
      float e = __expf(v[j] - m);
      v[j] = e;
      s += e;
    }
    float inv = 1.f / s;
    _Float16 row[16];
#pragma unroll
    for (int j = 0; j < 9; ++j) {
      _Float16 hv = (_Float16)(v[j] * inv);
      row[j] = hv;
      affT_s[j][tid] = hv;
    }
#pragma unroll
    for (int j = 9; j < 16; ++j) row[j] = (_Float16)0;
    _Float16* dst = aff_g + ((size_t)(b * NTOK + n) * 64 + tid) * 16;
    *(v8h*)dst = *(const v8h*)&row[0];
    *(v8h*)(dst + 8) = *(const v8h*)&row[8];
  } else {
    for (int i = tid - 64; i < 7 * 64; i += 64)
      affT_s[9 + (i >> 6)][i & 63] = (_Float16)0;
  }
  __syncthreads();

  if (tid < 9) {
    float s = 0.f;
    for (int p = 0; p < 64; ++p) s += (float)affT_s[tid][p];
    asum_pre[(b * NTOK + n) * 16 + tid] = s;
  }
  __syncthreads();

  // phase 3: sfk(16x256) = affT(16x64) @ pix(64x256); 4 waves x 4 N-tiles
  {
    float* dst = sfk + ((size_t)(b * NTOK + n) * 16) * 256;
    for (int t4 = 0; t4 < 4; ++t4) {
      int n0 = (wv * 4 + t4) * 16;
      v8f acc = vzero8f();
#pragma unroll
      for (int k0 = 0; k0 < 64; k0 += 32) {
        v16h a = load_a(&affT_s[0][0] + k0, 64, lane);
        v16h bm = load_bt(&pixT_s[n0][0] + k0, 64, lane);
        acc = wmma16(a, bm, acc);
      }
      int col = n0 + (lane & 15);
      int rb = (lane >> 4) << 3;
#pragma unroll
      for (int i = 0; i < 8; ++i) dst[(rb + i) * 256 + col] = acc[i];
    }
  }
}

// ---------------- K3: fold9 + normalize -> sf (b, token, c) f16 -----------
__global__ void __launch_bounds__(256) k_fold(const float* sfk,
                                              const float* asum_pre,
                                              _Float16* sf_hwc) {
  int blk = blockIdx.x;
  int t = blk & 255, b = blk >> 8;
  int y = t >> 4, xw = t & 15;
  int c = threadIdx.x;
  float acc = 0.f, asum = 0.f;
#pragma unroll
  for (int dy = 0; dy < 3; ++dy)
#pragma unroll
    for (int dx = 0; dx < 3; ++dx) {
      int ys = y + 1 - dy, xs = xw + 1 - dx;
      if ((unsigned)ys < 16u && (unsigned)xs < 16u) {
        int nn = ys * 16 + xs, j = dy * 3 + dx;
        acc += sfk[((size_t)(b * NTOK + nn) * 16 + j) * 256 + c];
        asum += asum_pre[(b * NTOK + nn) * 16 + j];
      }
    }
  sf_hwc[(size_t)blk * CC + c] = (_Float16)(acc / (asum + 1e-12f));
}

// ---------------- K4a: qkv = sf @ w_qkv^T (dual output layouts) -----------
__global__ void __launch_bounds__(256) k_qkv(const _Float16* sf_hwc,
                                             const _Float16* wqkv_h,
                                             _Float16* qkv_nt, _Float16* qkv_on) {
  int blk = blockIdx.x;
  int b = blk / 96;
  int tile = (blk % 96) * 8 + (threadIdx.x >> 5);   // 0..767
  int lane = threadIdx.x & 31;
  int mt = tile / 48, nt = tile % 48;               // token tile, out-chan tile
  const _Float16* arow = sf_hwc + ((size_t)b * NTOK + mt * 16) * CC;
  const _Float16* brow = wqkv_h + (size_t)nt * 16 * CC;
  v8f acc = vzero8f();
#pragma unroll
  for (int k0 = 0; k0 < 256; k0 += 32)
    acc = wmma16(load_a(arow + k0, CC, lane), load_bt(brow + k0, CC, lane), acc);
  int col = nt * 16 + (lane & 15);
  int rb = mt * 16 + ((lane >> 4) << 3);
#pragma unroll
  for (int i = 0; i < 8; ++i) {
    _Float16 hv = (_Float16)acc[i];
    qkv_nt[((size_t)b * NTOK + rb + i) * 768 + col] = hv;
    qkv_on[((size_t)b * 768 + col) * NTOK + rb + i] = hv;
  }
}

// ---------------- K4b: per-head attention (S=q^T k, softmax, o=attn v^T) --
__global__ void __launch_bounds__(256) k_attn(const _Float16* qkv_nt,
                                              const _Float16* qkv_on,
                                              _Float16* o_nt) {
  __shared__ float S_s[32][256];
  __shared__ __align__(16) _Float16 attn_s[32][256];
  __shared__ float po_s[8][256];
  __shared__ float red_s[32][8];

  int blk = blockIdx.x;
  int ib = blk & 7, h = (blk >> 3) & 7, b = blk >> 6;
  int i0 = ib * 32;
  int tid = threadIdx.x, lane = tid & 31, wv = tid >> 5;
  const _Float16* qbase = qkv_nt + (size_t)b * NTOK * 768;

  // phase 1: S(32x256), K=32 -> single WMMA per tile; 8 waves x 4 tiles
  for (int t = 0; t < 4; ++t) {
    int idx = wv * 4 + t;
    int mt = idx >> 4, nt = idx & 15;
    const _Float16* arow = qbase + (size_t)(i0 + mt * 16) * 768 + h * 96;       // q^T
    const _Float16* brow = qbase + (size_t)(nt * 16) * 768 + h * 96 + 32;       // k^T
    v8f acc = vzero8f();
    acc = wmma16(load_a(arow, 768, lane), load_bt(brow, 768, lane), acc);
    int col = nt * 16 + (lane & 15);
    int rb = mt * 16 + ((lane >> 4) << 3);
#pragma unroll
    for (int i = 0; i < 8; ++i) S_s[rb + i][col] = acc[i];
  }
  __syncthreads();

  // phase 2: row softmax (scaled), 8 threads per row
  {
    int r = tid >> 3, part = tid & 7, j0 = part * 32;
    float m = -1e30f;
    for (int j = 0; j < 32; ++j) m = fmaxf(m, S_s[r][j0 + j] * ATTN_SC);
    red_s[r][part] = m;
    __syncthreads();
    float rm = red_s[r][0];
    for (int p = 1; p < 8; ++p) rm = fmaxf(rm, red_s[r][p]);
    __syncthreads();
    float s = 0.f;
    for (int j = 0; j < 32; ++j) s += __expf(S_s[r][j0 + j] * ATTN_SC - rm);
    red_s[r][part] = s;
    __syncthreads();
    float rs = 0.f;
    for (int p = 0; p < 8; ++p) rs += red_s[r][p];
    float inv = 1.f / rs;
    for (int j = 0; j < 32; ++j)
      attn_s[r][j0 + j] = (_Float16)(__expf(S_s[r][j0 + j] * ATTN_SC - rm) * inv);
  }
  __syncthreads();

  // phase 3: oT(32x32) = attn(32x256) @ v^T ; K split over wave pairs
  {
    int t = wv & 3, kh = wv >> 2;
    int mt = t >> 1, nt = t & 1;
    const _Float16* vbase =
        qkv_on + ((size_t)b * 768 + h * 96 + 64 + nt * 16) * NTOK;  // Bt = v(d,j)
    v8f acc = vzero8f();
#pragma unroll
    for (int ks = 0; ks < 4; ++ks) {
      int k0 = kh * 128 + ks * 32;
      acc = wmma16(load_a(&attn_s[mt * 16][0] + k0, 256, lane),
                   load_bt(vbase + k0, NTOK, lane), acc);
    }
    int col = lane & 15;
    int rb = (lane >> 4) << 3;
#pragma unroll
    for (int i = 0; i < 8; ++i) po_s[wv][(rb + i) * 16 + col] = acc[i];
  }
  __syncthreads();
  for (int idx = tid; idx < 1024; idx += 256) {
    int t = idx >> 8, rem = idx & 255;
    int lr = rem >> 4, col = rem & 15;
    int mt = t >> 1, nt = t & 1;
    float v = po_s[t][rem] + po_s[t + 4][rem];
    o_nt[((size_t)b * NTOK + i0 + mt * 16 + lr) * CC + h * 32 + nt * 16 + col] =
        (_Float16)v;
  }
}

// ---------------- K4c: depthwise 3x3 PE on v + add into o ----------------
__global__ void __launch_bounds__(256) k_pe(const _Float16* qkv_on,
                                            const _Float16* o_nt,
                                            const float* w_pe,
                                            _Float16* onpe_nt) {
  int blk = blockIdx.x;
  int t = blk & 255, b = blk >> 8;
  int y = t >> 4, xw = t & 15;
  int c = threadIdx.x;
  int r = (c >> 5) * 96 + 64 + (c & 31);            // v row for channel c
  const _Float16* vrow = qkv_on + ((size_t)b * 768 + r) * NTOK;
  float acc = 0.f;
#pragma unroll
  for (int dy = 0; dy < 3; ++dy)
#pragma unroll
    for (int dx = 0; dx < 3; ++dx) {
      int ys = y + dy - 1, xs = xw + dx - 1;
      if ((unsigned)ys < 16u && (unsigned)xs < 16u)
        acc += w_pe[c * 9 + dy * 3 + dx] * (float)vrow[ys * 16 + xs];
    }
  size_t o = ((size_t)b * NTOK + t) * CC + c;
  onpe_nt[o] = (_Float16)((float)o_nt[o] + acc);
}

// ---------------- K4d: proj -> sf2 (b, token, c_out) f16 ------------------
__global__ void __launch_bounds__(256) k_proj(const _Float16* onpe_nt,
                                              const _Float16* wproj_h,
                                              _Float16* sf2_hwc) {
  int blk = blockIdx.x;
  int b = blk >> 5;
  int tile = (blk & 31) * 8 + (threadIdx.x >> 5);   // 0..255
  int lane = threadIdx.x & 31;
  int mt = tile >> 4, nt = tile & 15;
  const _Float16* arow = onpe_nt + ((size_t)b * NTOK + mt * 16) * CC;
  const _Float16* brow = wproj_h + (size_t)nt * 16 * CC;
  v8f acc = vzero8f();
#pragma unroll
  for (int k0 = 0; k0 < 256; k0 += 32)
    acc = wmma16(load_a(arow + k0, CC, lane), load_bt(brow + k0, CC, lane), acc);
  int col = nt * 16 + (lane & 15);
  int rb = mt * 16 + ((lane >> 4) << 3);
#pragma unroll
  for (int i = 0; i < 8; ++i)
    sf2_hwc[((size_t)b * NTOK + rb + i) * CC + col] = (_Float16)acc[i];
}

// ---------------- K5: scatter OutT(Cx64) = sf9b(Cx9) @ aff^T --------------
__global__ void __launch_bounds__(256) k_scatter(const _Float16* sf2_hwc,
                                                 const _Float16* aff_g,
                                                 float* out) {
  int blk = blockIdx.x;
  int n = blk & 255, b = blk >> 8;
  int yy = n >> 4, xx = n & 15;
  int tid = threadIdx.x, lane = tid & 31, wv = tid >> 5;

  int offj[9];                                      // uniform per block
#pragma unroll
  for (int j = 0; j < 9; ++j) {
    int y2 = yy + j / 3 - 1, x2 = xx + (j % 3) - 1;
    offj[j] = ((unsigned)y2 < 16u && (unsigned)x2 < 16u) ? (y2 * 16 + x2) * CC : -1;
  }
  const _Float16* sfbase = sf2_hwc + (size_t)b * NTOK * CC;
  float* obase = out + (size_t)b * CC * 16384;

  for (int q = 0; q < 8; ++q) {
    int idx = wv * 8 + q;                           // 16 c-tiles x 4 p-tiles
    int mt = idx >> 2, nt = idx & 3;
    int c = mt * 16 + (lane & 15);
    v16h a = vzero16h();                            // A row c over j (K padded 32)
    if (lane < 16) {
#pragma unroll
      for (int i2 = 0; i2 < 9; ++i2)
        if (offj[i2] >= 0) a[i2] = sfbase[offj[i2] + c];
    } else {
      if (offj[8] >= 0) a[0] = sfbase[offj[8] + c];
    }
    v16h bm;
    if (lane < 16) {                                // Bt = aff (p,16), K 16..31 = 0
      const _Float16* p =
          aff_g + ((size_t)(b * NTOK + n) * 64 + nt * 16 + lane) * 16;
      bm = pack16(*(const v8h*)p, *(const v8h*)(p + 8));
    } else {
      bm = vzero16h();
    }
    v8f acc = wmma16(a, bm, vzero8f());
    int p = nt * 16 + (lane & 15);
    int rpix = p >> 3, spix = p & 7;
    int cb = mt * 16 + ((lane >> 4) << 3);
#pragma unroll
    for (int i = 0; i < 8; ++i)
      obase[(size_t)(cb + i) * 16384 + (yy * 8 + rpix) * 128 + xx * 8 + spix] =
          acc[i];
  }
}

}  // namespace

extern "C" void kernel_launch(void* const* d_in, const int* in_sizes, int n_in,
                              void* d_out, int out_size, void* d_ws, size_t ws_size,
                              hipStream_t stream) {
  (void)in_sizes; (void)n_in; (void)out_size; (void)ws_size;
  const float* x      = (const float*)d_in[0];
  const float* w_qkv  = (const float*)d_in[1];
  const float* w_pe   = (const float*)d_in[2];
  const float* w_proj = (const float*)d_in[3];
  float* out = (float*)d_out;

  char* ws = (char*)d_ws;
  size_t off = 0;
  auto take = [&](size_t bytes) -> char* {
    char* p = ws + off;
    off += (bytes + 255) & ~(size_t)255;
    return p;
  };
  _Float16* wqkv_h     = (_Float16*)take((size_t)768 * 256 * 2);
  _Float16* wproj_h    = (_Float16*)take((size_t)256 * 256 * 2);
  _Float16* stoken_hwc = (_Float16*)take((size_t)BB * 16 * 16 * CC * 2);
  _Float16* aff_g      = (_Float16*)take((size_t)BB * NTOK * 64 * 16 * 2);
  float*    asum_pre   = (float*)take((size_t)BB * NTOK * 16 * 4);
  float*    sfk        = (float*)take((size_t)BB * NTOK * 16 * CC * 4);
  _Float16* sf_hwc     = (_Float16*)take((size_t)BB * NTOK * CC * 2);
  _Float16* qkv_nt     = (_Float16*)take((size_t)BB * NTOK * 768 * 2);
  _Float16* qkv_on     = (_Float16*)take((size_t)BB * 768 * NTOK * 2);
  _Float16* o_nt       = (_Float16*)take((size_t)BB * NTOK * CC * 2);
  _Float16* onpe_nt    = (_Float16*)take((size_t)BB * NTOK * CC * 2);
  _Float16* sf2_hwc    = (_Float16*)take((size_t)BB * NTOK * CC * 2);

  k_cvt_w<<<768, 256, 0, stream>>>(w_qkv, w_proj, wqkv_h, wproj_h);
  k_pool<<<BB * CC * HH, 128, 0, stream>>>(x, stoken_hwc);
  k_aff<<<BB * NTOK, 128, 0, stream>>>(x, stoken_hwc, aff_g, asum_pre, sfk);
  k_fold<<<BB * NTOK, 256, 0, stream>>>(sfk, asum_pre, sf_hwc);
  k_qkv<<<BB * 96, 256, 0, stream>>>(sf_hwc, wqkv_h, qkv_nt, qkv_on);
  k_attn<<<BB * NHEAD * 8, 256, 0, stream>>>(qkv_nt, qkv_on, o_nt);
  k_pe<<<BB * NTOK, 256, 0, stream>>>(qkv_on, o_nt, w_pe, onpe_nt);
  k_proj<<<BB * 32, 256, 0, stream>>>(onpe_nt, wproj_h, sf2_hwc);
  k_scatter<<<BB * NTOK, 256, 0, stream>>>(sf2_hwc, aff_g, out);
}
